// MaskFormer3D_15762529976903
// MI455X (gfx1250) — compile-verified, matrix operand order checked by reading, not run on hardware
//
#include <hip/hip_runtime.h>
#include <hip/hip_bf16.h>

// ---------------------------------------------------------------------------
// MaskFormer3D-style pipeline for gfx1250 (CDNA5, wave32, WMMA).
// Token-major activations (B, cols, Kpad); WMMA fragments are contiguous 16B
// runs. GEMM uses double-buffered GLOBAL_LOAD_ASYNC_TO_LDS_B128 (ASYNCcnt)
// to overlap global->LDS staging with v_wmma_f32_16x16x32_f16 bursts.
// ---------------------------------------------------------------------------

typedef __attribute__((ext_vector_type(16))) _Float16 v16h;
typedef __attribute__((ext_vector_type(8)))  _Float16 h8;
typedef __attribute__((ext_vector_type(8)))  float    v8f;

#define Bn  4
#define Nn  16384
#define NP1 1024
#define NP2 512
#define NSAMP 64
#define DEMB 256

union Frag { v16h v; h8 h[2]; };

// ======================= Weight prepack ====================================
// Apack fragment layout: [kt][mt][lane][16 halves], matching the ISA's
// 16-bit A-matrix 16x32 VGPR striping.  One thread per fragment.
__global__ void prepack_w_kernel(const float* __restrict__ W, _Float16* __restrict__ Apack,
                                 int M, int K, int Kp, int M16) {
    int t = blockIdx.x * blockDim.x + threadIdx.x;
    int total = (Kp / 32) * M16 * 32;
    if (t >= total) return;
    int lane = t & 31;
    int r = t >> 5;
    int mt = r % M16;
    int kt = r / M16;
    int m = mt * 16 + (lane & 15);
    int koff = kt * 32 + ((lane < 16) ? 0 : 8);
    h8 lo, hi;
    #pragma unroll
    for (int e = 0; e < 8; e++) {
        int k0 = koff + e;
        int k1 = koff + 16 + e;
        lo[e] = (k0 < K) ? (_Float16)W[(size_t)m * K + k0] : (_Float16)0.f;
        hi[e] = (k1 < K) ? (_Float16)W[(size_t)m * K + k1] : (_Float16)0.f;
    }
    h8* op = (h8*)(Apack + (size_t)t * 16);
    op[0] = lo; op[1] = hi;
}

// ======================= WMMA GEMM =========================================
// Y[b, col, m] = sum_k W[m,k] * X[b, col, k]  (+ bias[m])
// Apack: prepacked f16 fragments.  X: f16 (B, cols, Kp).  Y: OutT (B, cols, M).
template <typename OutT>
__global__ __launch_bounds__(256)
void gemm_wmma_kernel(const _Float16* __restrict__ Apack, const _Float16* __restrict__ X,
                      OutT* __restrict__ Y, const float* __restrict__ bias,
                      int M, int Kp, int cols)
{
    __shared__ h8 Bs[2][8][32][2];   // double-buffered: 2 x 8 n-subtiles x 32 lanes x 32B

    const int b  = blockIdx.z;
    const int m0 = blockIdx.y * 128;
    const int c0 = blockIdx.x * 128;
    const int M16 = M >> 4;
    const _Float16* Xb = X + (size_t)b * cols * Kp;
    OutT* Yb = Y + (size_t)b * cols * M;

    const int tid  = threadIdx.x;
    const int lane = tid & 31;
    const int wave = tid >> 5;     // 0..7
    const int wm   = wave & 3;     // M offset wm*32
    const int wn   = wave >> 2;    // col offset wn*64
    const int rowlane = lane & 15;

    v8f acc[2][4];
    for (int i = 0; i < 2; i++)
        for (int j = 0; j < 4; j++)
            for (int r = 0; r < 8; r++) acc[i][j][r] = 0.f;

    // ---- this thread's staging duty: one B fragment per buffer slot
    const int snt = tid >> 5;
    const int sl  = tid & 31;
    const int scol = c0 + snt * 16 + (sl & 15);
    const _Float16* sgbase = Xb + (size_t)scol * Kp + ((sl < 16) ? 0 : 8);
    const unsigned lds0 = (unsigned)(size_t)&Bs[0][snt][sl][0];
    const unsigned lds1 = (unsigned)(size_t)&Bs[1][snt][sl][0];

    // pre-zero OOB fragment slots (never touched by async ops)
    if (scol >= cols) {
        h8 z;
        #pragma unroll
        for (int e = 0; e < 8; e++) z[e] = (_Float16)0.f;
        Bs[0][snt][sl][0] = z; Bs[0][snt][sl][1] = z;
        Bs[1][snt][sl][0] = z; Bs[1][snt][sl][1] = z;
    }

    const int KT = Kp >> 5;
    // ---- prologue: async-copy kt=0 fragments into buffer 0
    if (scol < cols) {
        unsigned long long ga = (unsigned long long)(size_t)sgbase;
        asm volatile("global_load_async_to_lds_b128 %0, %1, off"
                     :: "v"(lds0), "v"(ga) : "memory");
        asm volatile("global_load_async_to_lds_b128 %0, %1, off"
                     :: "v"(lds0 + 16u), "v"(ga + 32ull) : "memory");
    }

    for (int kt = 0; kt < KT; kt++) {
        // ---- issue async copies for the NEXT K-tile, then wait for current
        if (kt + 1 < KT) {
            if (scol < cols) {
                unsigned long long ga = (unsigned long long)(size_t)(sgbase + (size_t)(kt + 1) * 32);
                unsigned ldsn = ((kt + 1) & 1) ? lds1 : lds0;
                asm volatile("global_load_async_to_lds_b128 %0, %1, off"
                             :: "v"(ldsn), "v"(ga) : "memory");
                asm volatile("global_load_async_to_lds_b128 %0, %1, off"
                             :: "v"(ldsn + 16u), "v"(ga + 32ull) : "memory");
            }
            // 2 newest (next tile) may fly; 2 oldest (current tile) must be done
            asm volatile("s_wait_asynccnt 0x2" ::: "memory");
        } else {
            asm volatile("s_wait_asynccnt 0x0" ::: "memory");
        }
        __syncthreads();

        const int cur = kt & 1;
        // ---- A fragments straight from prepacked global (coalesced, L0-hot)
        Frag a[2];
        #pragma unroll
        for (int i = 0; i < 2; i++) {
            int mt = (m0 >> 4) + wm * 2 + i;
            if (mt < M16) {
                const h8* ap = (const h8*)(Apack + ((size_t)(kt * M16 + mt) * 32 + lane) * 16);
                a[i].h[0] = ap[0];
                a[i].h[1] = ap[1];
            } else {
                #pragma unroll
                for (int e = 0; e < 8; e++) { a[i].h[0][e] = (_Float16)0.f; a[i].h[1][e] = (_Float16)0.f; }
            }
        }
        // ---- B fragments from LDS (2x ds_load_b128 each)
        Frag bf[4];
        #pragma unroll
        for (int j = 0; j < 4; j++) {
            bf[j].h[0] = Bs[cur][wn * 4 + j][lane][0];
            bf[j].h[1] = Bs[cur][wn * 4 + j][lane][1];
        }
        #pragma unroll
        for (int i = 0; i < 2; i++)
            #pragma unroll
            for (int j = 0; j < 4; j++)
                acc[i][j] = __builtin_amdgcn_wmma_f32_16x16x32_f16(
                    false, a[i].v, false, bf[j].v, (short)0, acc[i][j], false, false);
        __syncthreads();   // protect cur buffer before next iteration overwrites it
    }

    // ---- store: 8 consecutive channels per lane -> vectorizable stores
    const int rbase = (lane < 16) ? 0 : 8;
    #pragma unroll
    for (int i = 0; i < 2; i++) {
        const int row0 = m0 + wm * 32 + i * 16 + rbase;
        if (row0 >= M) continue;
        #pragma unroll
        for (int j = 0; j < 4; j++) {
            const int colg = c0 + wn * 64 + j * 16 + rowlane;
            if (colg < cols) {
                OutT* yp = Yb + (size_t)colg * M + row0;
                #pragma unroll
                for (int r = 0; r < 8; r++) {
                    float v = acc[i][j][r];
                    if (bias) v += bias[row0 + r];
                    yp[r] = (OutT)v;
                }
            }
        }
    }
}

// ======================= Elementwise / misc ================================
__global__ void f32_to_f16_kernel(const float* x, _Float16* y, size_t n) {
    size_t i = (size_t)blockIdx.x * blockDim.x + threadIdx.x;
    if (i < n) y[i] = (_Float16)x[i];
}

// ======================= FPS ===============================================
__global__ __launch_bounds__(1024)
void fps_kernel(const float* xyz, int n, int npoint, int* idx_out) {
    const int b = blockIdx.x;
    const int tid = threadIdx.x;
    const int T = blockDim.x;           // 1024
    const int per = (n + T - 1) / T;    // <= 16
    const float* p = xyz + (size_t)b * n * 3;

    __shared__ float rv[1024];
    __shared__ int   ri[1024];
    __shared__ float cx, cy, cz;

    float dists[16];
    for (int i = 0; i < 16; i++) dists[i] = 1e10f;

    int far = 0;
    for (int it = 0; it < npoint; it++) {
        if (tid == 0) {
            idx_out[b * npoint + it] = far;
            cx = p[(size_t)far * 3]; cy = p[(size_t)far * 3 + 1]; cz = p[(size_t)far * 3 + 2];
        }
        __syncthreads();
        float bestv = -1.f; int besti = 0;
        for (int i = 0; i < per; i++) {
            int j = tid * per + i;
            if (j < n) {
                float dx = p[(size_t)j * 3] - cx;
                float dy = p[(size_t)j * 3 + 1] - cy;
                float dz = p[(size_t)j * 3 + 2] - cz;
                float d = dx * dx + dy * dy + dz * dz;
                if (d < dists[i]) dists[i] = d;
                if (dists[i] > bestv) { bestv = dists[i]; besti = j; }
            }
        }
        rv[tid] = bestv; ri[tid] = besti;
        __syncthreads();
        for (int s = T / 2; s > 0; s >>= 1) {
            if (tid < s) {
                if (rv[tid + s] > rv[tid] ||
                    (rv[tid + s] == rv[tid] && ri[tid + s] < ri[tid])) {
                    rv[tid] = rv[tid + s]; ri[tid] = ri[tid + s];
                }
            }
            __syncthreads();
        }
        far = ri[0];
    }
}

__global__ void gather_xyz_kernel(const float* xyz, const int* idx, float* out, int n, int np) {
    int i = blockIdx.x * blockDim.x + threadIdx.x;
    if (i >= Bn * np * 3) return;
    int c = i % 3; int s = (i / 3) % np; int b = i / (3 * np);
    out[i] = xyz[((size_t)b * n + idx[b * np + s]) * 3 + c];
}

// ======================= Ball query (wave32 per centroid) ==================
__global__ void ball_query_kernel(const float* xyz, const float* new_xyz,
                                  int n, int np, float r2, int ns, int* out) {
    int gw = (blockIdx.x * blockDim.x + threadIdx.x) >> 5;
    int lane = threadIdx.x & 31;
    if (gw >= Bn * np) return;
    int b = gw / np, s = gw % np;
    const float* p = xyz + (size_t)b * n * 3;
    float cx = new_xyz[((size_t)b * np + s) * 3];
    float cy = new_xyz[((size_t)b * np + s) * 3 + 1];
    float cz = new_xyz[((size_t)b * np + s) * 3 + 2];
    int* o = out + ((size_t)b * np + s) * ns;

    int cnt = 0;
    for (int base = 0; base < n && cnt < ns; base += 32) {
        int j = base + lane;
        bool hit = false;
        if (j < n) {
            float dx = p[(size_t)j * 3] - cx;
            float dy = p[(size_t)j * 3 + 1] - cy;
            float dz = p[(size_t)j * 3 + 2] - cz;
            hit = (dx * dx + dy * dy + dz * dz) < r2;
        }
        unsigned mask = __builtin_amdgcn_ballot_w32(hit);
        while (mask && cnt < ns) {
            int bpos = __ffs((int)mask) - 1;
            if (lane == 0) o[cnt] = base + bpos;
            cnt++;
            mask &= mask - 1;
        }
    }
    if (lane == 0) {
        int first = (cnt > 0) ? o[0] : 0;
        for (int k = cnt; k < ns; k++) o[k] = first;
    }
}

// ======================= Grouped-feature build (token-major) ===============
// Xg[b, s*ns+k, c]: c<3 -> xyz[idx]-new_xyz ; c<3+C -> feats[b,idx,c-3] ; else 0
__global__ void build_group_kernel(const float* xyz, const float* new_xyz,
                                   const float* feats, const int* idx,
                                   int n, int np, int ns, int C, int Kp, _Float16* Xg) {
    size_t tid = (size_t)blockIdx.x * blockDim.x + threadIdx.x;
    size_t total = (size_t)Bn * np * ns * Kp;
    if (tid >= total) return;
    int c = tid % Kp; size_t t = tid / Kp;
    int k = t % ns; t /= ns;
    int s = t % np; int b = t / np;
    float v = 0.f;
    if (c < 3 + C) {
        int id = idx[((size_t)b * np + s) * ns + k];
        if (c < 3) v = xyz[((size_t)b * n + id) * 3 + c] - new_xyz[((size_t)b * np + s) * 3 + c];
        else       v = feats[((size_t)b * n + id) * C + (c - 3)];
    }
    Xg[tid] = (_Float16)v;
}

// ======================= GroupNorm (G=4), token-major ======================
__global__ void gn_stats_kernel(const _Float16* Y, int M, int cols, float* stats) {
    int b = blockIdx.z, g = blockIdx.y;
    int Mg = M / 4;
    size_t nelem = (size_t)Mg * cols;
    const _Float16* p = Y + (size_t)b * cols * M + g * Mg;
    float s = 0.f, ss = 0.f;
    size_t stride = (size_t)gridDim.x * blockDim.x;
    for (size_t i = (size_t)blockIdx.x * blockDim.x + threadIdx.x; i < nelem; i += stride) {
        size_t col = i / Mg; int cc = (int)(i % Mg);
        float v = (float)p[col * M + cc]; s += v; ss += v * v;
    }
    __shared__ float rs[256], rss[256];
    rs[threadIdx.x] = s; rss[threadIdx.x] = ss;
    __syncthreads();
    for (int st = 128; st > 0; st >>= 1) {
        if (threadIdx.x < st) { rs[threadIdx.x] += rs[threadIdx.x + st]; rss[threadIdx.x] += rss[threadIdx.x + st]; }
        __syncthreads();
    }
    if (threadIdx.x == 0) {
        atomicAdd(&stats[(b * 4 + g) * 2], rs[0]);
        atomicAdd(&stats[(b * 4 + g) * 2 + 1], rss[0]);
    }
}

__global__ void gn_apply_kernel(const _Float16* Y, const float* stats,
                                const float* gamma, const float* beta,
                                int M, int cols, float inv_ne,
                                _Float16* outH, float* outF) {
    size_t tid = (size_t)blockIdx.x * blockDim.x + threadIdx.x;
    size_t total = (size_t)Bn * cols * M;
    if (tid >= total) return;
    int m = tid % M;
    int b = tid / ((size_t)cols * M);
    int g = m / (M / 4);
    float sum = stats[(b * 4 + g) * 2];
    float ssum = stats[(b * 4 + g) * 2 + 1];
    float mu = sum * inv_ne;
    float var = ssum * inv_ne - mu * mu;
    float rstd = rsqrtf(var + 1e-5f);
    float v = ((float)Y[tid] - mu) * rstd * gamma[m] + beta[m];
    v = fmaxf(v, 0.f);   // all GN uses are followed by relu
    if (outH) outH[tid] = (_Float16)v;
    if (outF) outF[tid] = v;
}

// ======================= Max over samples (token-major) ====================
__global__ void max_ns_kernel(const _Float16* H, int C, int np, int ns,
                              int Ctot, int coff, float* outF, _Float16* outH) {
    int i = blockIdx.x * blockDim.x + threadIdx.x;
    if (i >= Bn * np * C) return;
    int c = i % C; int s = (i / C) % np; int b = i / (C * np);
    const _Float16* p = H + ((size_t)(b * np + s) * ns) * C;
    float m = -1e30f;
    for (int k = 0; k < ns; k++) { float v = (float)p[(size_t)k * C + c]; if (v > m) m = v; }
    size_t o = ((size_t)b * np + s) * Ctot + coff + c;
    outF[o] = m;
    if (outH) outH[o] = (_Float16)m;
}

// ======================= kNN-3 + interpolation =============================
__global__ void knn3_kernel(const float* unk, const float* kn, int n, int m,
                            int* idx3, float* w3) {
    int i = blockIdx.x * blockDim.x + threadIdx.x;
    if (i >= Bn * n) return;
    int u = i % n; int b = i / n;
    float ux = unk[((size_t)b * n + u) * 3];
    float uy = unk[((size_t)b * n + u) * 3 + 1];
    float uz = unk[((size_t)b * n + u) * 3 + 2];
    const float* kp = kn + (size_t)b * m * 3;
    float d0 = 1e30f, d1 = 1e30f, dd2 = 1e30f;
    int i0 = 0, i1 = 0, i2 = 0;
    for (int j = 0; j < m; j++) {
        float dx = kp[(size_t)j * 3] - ux;
        float dy = kp[(size_t)j * 3 + 1] - uy;
        float dz = kp[(size_t)j * 3 + 2] - uz;
        float d = dx * dx + dy * dy + dz * dz;
        if (d < d0)      { dd2 = d1; i2 = i1; d1 = d0; i1 = i0; d0 = d; i0 = j; }
        else if (d < d1) { dd2 = d1; i2 = i1; d1 = d; i1 = j; }
        else if (d < dd2){ dd2 = d; i2 = j; }
    }
    float w0 = 1.f / (d0 + 1e-8f), w1 = 1.f / (d1 + 1e-8f), w2 = 1.f / (dd2 + 1e-8f);
    float ws = w0 + w1 + w2;
    size_t o = ((size_t)b * n + u) * 3;
    idx3[o] = i0; idx3[o + 1] = i1; idx3[o + 2] = i2;
    w3[o] = w0 / ws; w3[o + 1] = w1 / ws; w3[o + 2] = w2 / ws;
}

// token-major: Xg[b, u, c]; kn_feats (B, m, Ck); unk_feats (B, n, Cu)
__global__ void interp_concat_kernel(const float* kn_feats, int Ck, int m,
                                     const float* unk_feats, int Cu, int n,
                                     const int* idx3, const float* w3,
                                     int Kp, _Float16* Xg) {
    size_t tid = (size_t)blockIdx.x * blockDim.x + threadIdx.x;
    size_t total = (size_t)Bn * n * Kp;
    if (tid >= total) return;
    int c = tid % Kp; size_t t = tid / Kp;
    int u = t % n; int b = t / n;
    float v = 0.f;
    if (c < Ck) {
        const int* id = &idx3[((size_t)b * n + u) * 3];
        const float* w = &w3[((size_t)b * n + u) * 3];
        const float* f = kn_feats + (size_t)b * m * Ck;
        v = w[0] * f[(size_t)id[0] * Ck + c] + w[1] * f[(size_t)id[1] * Ck + c]
          + w[2] * f[(size_t)id[2] * Ck + c];
    } else if (c < Ck + Cu) {
        v = unk_feats[((size_t)b * n + u) * Cu + (c - Ck)];
    }
    Xg[tid] = (_Float16)v;
}

// ======================= Decoder helpers ===================================
__global__ void bcast_query_kernel(const float* q, float* tgt) {
    int i = blockIdx.x * blockDim.x + threadIdx.x;
    if (i >= 16 * DEMB) return;
    for (int b = 0; b < Bn; b++) tgt[b * 16 * DEMB + i] = q[i];
}

__global__ void linear_kernel(const float* X, const float* W, const float* bias,
                              float* Y, int L, int Din, int Dout, int relu) {
    int i = blockIdx.x * blockDim.x + threadIdx.x;
    if (i >= Bn * L * Dout) return;
    int o = i % Dout; int r = (i / Dout) % L; int b = i / (Dout * L);
    const float* x = X + ((size_t)b * L + r) * Din;
    const float* w = W + (size_t)o * Din;
    float acc = bias ? bias[o] : 0.f;
    for (int d = 0; d < Din; d++) acc += x[d] * w[d];
    if (relu) acc = fmaxf(acc, 0.f);
    Y[((size_t)b * L + r) * Dout + o] = acc;
}

// one wave32 per (b, head, query); K,V row-major (B, Lk, 256)
__global__ __launch_bounds__(32)
void mha_kernel(const float* Q, const float* K, const float* V,
                int Lq, int Lk, float* Out) {
    int gid = blockIdx.x;
    int q = gid % Lq; int h = (gid / Lq) % 8; int b = gid / (Lq * 8);
    int lane = threadIdx.x;
    __shared__ float sc[512];
    __shared__ float qv[32];
    __shared__ float red[32];

    qv[lane] = Q[((size_t)b * Lq + q) * DEMB + h * 32 + lane];
    __syncthreads();

    float lmax = -1e30f;
    for (int m = lane; m < Lk; m += 32) {
        const float* kp = K + ((size_t)b * Lk + m) * DEMB + h * 32;
        float s = 0.f;
        for (int d = 0; d < 32; d++) s += qv[d] * kp[d];
        s *= 0.17677669529663687f;  // 1/sqrt(32)
        sc[m] = s;
        if (s > lmax) lmax = s;
    }
    red[lane] = lmax; __syncthreads();
    for (int st = 16; st > 0; st >>= 1) { if (lane < st) red[lane] = fmaxf(red[lane], red[lane + st]); __syncthreads(); }
    float mx = red[0]; __syncthreads();
    float lsum = 0.f;
    for (int m = lane; m < Lk; m += 32) { float e = __expf(sc[m] - mx); sc[m] = e; lsum += e; }
    red[lane] = lsum; __syncthreads();
    for (int st = 16; st > 0; st >>= 1) { if (lane < st) red[lane] += red[lane + st]; __syncthreads(); }
    float inv = 1.f / red[0];
    __syncthreads();

    float acc = 0.f;
    for (int m = 0; m < Lk; m++) acc += sc[m] * V[((size_t)b * Lk + m) * DEMB + h * 32 + lane];
    Out[((size_t)b * Lq + q) * DEMB + h * 32 + lane] = acc * inv;
}

__global__ __launch_bounds__(256)
void add_ln_kernel(float* x, const float* res, const float* g, const float* bb) {
    int row = blockIdx.x; int t = threadIdx.x;
    __shared__ float sh[256];
    float v = x[(size_t)row * 256 + t] + res[(size_t)row * 256 + t];
    sh[t] = v; __syncthreads();
    for (int s = 128; s > 0; s >>= 1) { if (t < s) sh[t] += sh[t + s]; __syncthreads(); }
    float mu = sh[0] * (1.f / 256.f); __syncthreads();
    float d = v - mu; sh[t] = d * d; __syncthreads();
    for (int s = 128; s > 0; s >>= 1) { if (t < s) sh[t] += sh[t + s]; __syncthreads(); }
    float var = sh[0] * (1.f / 256.f);
    x[(size_t)row * 256 + t] = d * rsqrtf(var + 1e-5f) * g[t] + bb[t];
}

// ======================= Final logits + softmax ============================
// f0u (B, N, 64), s2 (B, 16, 64) token-major
__global__ __launch_bounds__(256)
void logits_softmax_kernel(const float* f0u, const float* s2, float* out, int n) {
    __shared__ float sn[16 * 64];
    __shared__ float rn[16];
    int b = blockIdx.y; int t = threadIdx.x;
    if (t < 16) {
        float s = 0.f;
        for (int c = 0; c < 64; c++) { float v = s2[((size_t)b * 16 + t) * 64 + c]; s += v * v; }
        rn[t] = 1.f / fmaxf(sqrtf(s), 1e-12f);
    }
    __syncthreads();
    for (int i = t; i < 1024; i += 256) sn[i] = s2[(size_t)b * 1024 + i] * rn[i >> 6];
    __syncthreads();
    int j = blockIdx.x * blockDim.x + t;
    if (j < n) {
        float f[64]; float nr = 0.f;
        for (int c = 0; c < 64; c++) { float v = f0u[((size_t)b * n + j) * 64 + c]; f[c] = v; nr += v * v; }
        nr = 1.f / fmaxf(sqrtf(nr), 1e-12f);
        float lg[16]; float mx = -1e30f;
        for (int k = 0; k < 16; k++) {
            float d = 0.f;
            for (int c = 0; c < 64; c++) d += f[c] * sn[k * 64 + c];
            d *= nr * 20.f;   // / 0.05
            lg[k] = d; if (d > mx) mx = d;
        }
        float ssum = 0.f;
        for (int k = 0; k < 16; k++) { float e = __expf(lg[k] - mx); lg[k] = e; ssum += e; }
        float inv = 1.f / ssum;
        for (int k = 0; k < 16; k++) out[((size_t)b * n + j) * 16 + k] = lg[k] * inv;
    }
}

// ======================= Host orchestration ================================
struct ConvGN { const float *W, *b, *g; };
struct AttnP  { const float *Wk, *Wo, *Wq, *Wv, *bk, *bo, *bq, *bv; };
struct LayerP {
    const float *W1, *W2, *b1, *b2;
    AttnP cross;
    const float *ln1b, *ln1g, *ln2b, *ln2g, *ln3b, *ln3g;
    AttnP slf;
};

static inline int round32(int x) { return (x + 31) & ~31; }

extern "C" void kernel_launch(void* const* d_in, const int* in_sizes, int n_in,
                              void* d_out, int out_size, void* d_ws, size_t ws_size,
                              hipStream_t stream) {
    (void)in_sizes; (void)n_in; (void)out_size; (void)ws_size;

    const float* pc = (const float*)d_in[0];          // (B,N,3)
    const float* point_feats = (const float*)d_in[1]; // (B,N,3) token-major already

    // ---- params (JAX pytree order: sorted dict keys) ----
    int pi = 2;
    auto nx = [&]() { return (const float*)d_in[pi++]; };
    LayerP dec[2];
    for (int l = 0; l < 2; l++) {
        LayerP& L = dec[l];
        L.W1 = nx(); L.W2 = nx(); L.b1 = nx(); L.b2 = nx();
        L.cross.Wk = nx(); L.cross.Wo = nx(); L.cross.Wq = nx(); L.cross.Wv = nx();
        L.cross.bk = nx(); L.cross.bo = nx(); L.cross.bq = nx(); L.cross.bv = nx();
        L.ln1b = nx(); L.ln1g = nx(); L.ln2b = nx(); L.ln2g = nx(); L.ln3b = nx(); L.ln3g = nx();
        L.slf.Wk = nx(); L.slf.Wo = nx(); L.slf.Wq = nx(); L.slf.Wv = nx();
        L.slf.bk = nx(); L.slf.bo = nx(); L.slf.bq = nx(); L.slf.bv = nx();
    }
    ConvGN fp1cg[3]; for (int i = 0; i < 3; i++) { fp1cg[i].W = nx(); fp1cg[i].b = nx(); fp1cg[i].g = nx(); }
    ConvGN fp2cg[2]; for (int i = 0; i < 2; i++) { fp2cg[i].W = nx(); fp2cg[i].b = nx(); fp2cg[i].g = nx(); }
    const float* inW = nx(); const float* inb = nx();
    ConvGN obj1cg; obj1cg.W = nx(); obj1cg.b = nx(); obj1cg.g = nx();
    const float* obj2W = nx(); const float* obj2b = nx();
    const float* queryP = nx();
    ConvGN sa1b0[3]; for (int i = 0; i < 3; i++) { sa1b0[i].W = nx(); sa1b0[i].b = nx(); sa1b0[i].g = nx(); }
    ConvGN sa1b1[3]; for (int i = 0; i < 3; i++) { sa1b1[i].W = nx(); sa1b1[i].b = nx(); sa1b1[i].g = nx(); }
    ConvGN sa2cg[3]; for (int i = 0; i < 3; i++) { sa2cg[i].W = nx(); sa2cg[i].b = nx(); sa2cg[i].g = nx(); }

    // ---- workspace ----
    size_t off = 0;
    auto alloc = [&](size_t bytes) -> void* {
        off = (off + 255) & ~(size_t)255;
        void* p = (char*)d_ws + off;
        off += bytes;
        return p;
    };
    int*      idxf1  = (int*)alloc((size_t)Bn * NP1 * 4);
    float*    xyz1   = (float*)alloc((size_t)Bn * NP1 * 3 * 4);
    int*      idxf2  = (int*)alloc((size_t)Bn * NP2 * 4);
    float*    xyz2   = (float*)alloc((size_t)Bn * NP2 * 3 * 4);
    int*      idxA   = (int*)alloc((size_t)Bn * NP1 * NSAMP * 4);
    int*      idxB   = (int*)alloc((size_t)Bn * NP1 * NSAMP * 4);
    int*      idxC   = (int*)alloc((size_t)Bn * NP2 * NSAMP * 4);
    float*    f1     = (float*)alloc((size_t)Bn * NP1 * 192 * 4);   // (B, NP1, 192)
    float*    f2     = (float*)alloc((size_t)Bn * NP2 * 256 * 4);   // (B, NP2, 256)
    _Float16* f2h    = (_Float16*)alloc((size_t)Bn * NP2 * 256 * 2);
    float*    f1u    = (float*)alloc((size_t)Bn * NP1 * 128 * 4);
    float*    f0u    = (float*)alloc((size_t)Bn * Nn * 64 * 4);
    int*      idx3a  = (int*)alloc((size_t)Bn * NP1 * 3 * 4);
    float*    w3a    = (float*)alloc((size_t)Bn * NP1 * 3 * 4);
    int*      idx3b  = (int*)alloc((size_t)Bn * Nn * 3 * 4);
    float*    w3b    = (float*)alloc((size_t)Bn * Nn * 3 * 4);
    _Float16* memh   = (_Float16*)alloc((size_t)Bn * NP2 * 256 * 2); // mem f16 (B,512,256)
    float*    KT_    = (float*)alloc((size_t)Bn * NP2 * 256 * 4);
    float*    VT_    = (float*)alloc((size_t)Bn * NP2 * 256 * 4);
    float*    tgt    = (float*)alloc((size_t)Bn * 16 * 256 * 4);
    float*    tq     = (float*)alloc((size_t)Bn * 16 * 256 * 4);
    float*    tk     = (float*)alloc((size_t)Bn * 16 * 256 * 4);
    float*    tv     = (float*)alloc((size_t)Bn * 16 * 256 * 4);
    float*    ta     = (float*)alloc((size_t)Bn * 16 * 256 * 4);
    float*    tb     = (float*)alloc((size_t)Bn * 16 * 256 * 4);
    _Float16* sTh    = (_Float16*)alloc((size_t)Bn * 16 * 256 * 2);
    float*    s2     = (float*)alloc((size_t)Bn * 16 * 64 * 4);
    float*    stats  = (float*)alloc(256);
    _Float16* Apack  = (_Float16*)alloc(262144);                     // max 256x448 f16 fragments
    const size_t BIG = (size_t)Bn * (NP2 * NSAMP) * 256 * 2;         // 64 MB
    _Float16* bufA = (_Float16*)alloc(BIG);
    _Float16* bufB = (_Float16*)alloc(BIG);
    _Float16* bufY = (_Float16*)alloc(BIG);

    // ---- helpers ----
    auto gemm_any = [&](const float* Wp, const _Float16* Xp, void* Yp, const float* biasp,
                        int M, int Kreal, int Kp, int cols, bool f16out) {
        int M16 = M / 16;
        int total = (Kp / 32) * M16 * 32;
        prepack_w_kernel<<<(total + 255) / 256, 256, 0, stream>>>(Wp, Apack, M, Kreal, Kp, M16);
        dim3 g((cols + 127) / 128, (M + 127) / 128, Bn);
        if (f16out)
            gemm_wmma_kernel<_Float16><<<g, 256, 0, stream>>>(Apack, Xp, (_Float16*)Yp, biasp, M, Kp, cols);
        else
            gemm_wmma_kernel<float><<<g, 256, 0, stream>>>(Apack, Xp, (float*)Yp, biasp, M, Kp, cols);
    };
    auto gnrun = [&](const _Float16* Y, const float* gamma, const float* beta,
                     int M, int cols, _Float16* outH, float* outF) {
        hipMemsetAsync(stats, 0, 256, stream);
        size_t ne = (size_t)(M / 4) * cols;
        int chunks = (int)((ne + 2047) / 2048);
        if (chunks > 512) chunks = 512;
        if (chunks < 1) chunks = 1;
        gn_stats_kernel<<<dim3(chunks, 4, Bn), 256, 0, stream>>>(Y, M, cols, stats);
        size_t tot = (size_t)Bn * cols * M;
        gn_apply_kernel<<<(unsigned)((tot + 255) / 256), 256, 0, stream>>>(
            Y, stats, gamma, beta, M, cols, 1.f / (float)ne, outH, outF);
    };
    auto run_stack = [&](const ConvGN* Ls, const int* dims, int nl, int cols,
                         _Float16* x0, _Float16* x1, float* finalF,
                         bool wantFinalH) -> _Float16* {
        _Float16* xin = x0; _Float16* xout = x1;
        for (int li = 0; li < nl; li++) {
            int K = dims[li], M = dims[li + 1];
            int Kp = round32(K);
            gemm_any(Ls[li].W, xin, bufY, nullptr, M, K, Kp, cols, true);
            bool last = (li == nl - 1);
            _Float16* oh = (!last || wantFinalH) ? xout : nullptr;
            float* of = last ? finalF : nullptr;
            gnrun(bufY, Ls[li].g, Ls[li].b, M, cols, oh, of);
            _Float16* t = xin; xin = xout; xout = t;
        }
        return xin;
    };

    // ===================== pipeline ========================================
    // ---- SA level 1 ----
    fps_kernel<<<Bn, 1024, 0, stream>>>(pc, Nn, NP1, idxf1);
    gather_xyz_kernel<<<(Bn * NP1 * 3 + 255) / 256, 256, 0, stream>>>(pc, idxf1, xyz1, Nn, NP1);
    {
        int blocks = (Bn * NP1 * 32 + 255) / 256;
        ball_query_kernel<<<blocks, 256, 0, stream>>>(pc, xyz1, Nn, NP1, 0.01f, NSAMP, idxA);
        ball_query_kernel<<<blocks, 256, 0, stream>>>(pc, xyz1, Nn, NP1, 0.04f, NSAMP, idxB);
    }
    const int cols1 = NP1 * NSAMP;  // 65536
    {   // branch 0: 6 -> 64 -> 64 -> 64
        size_t tot = (size_t)Bn * cols1 * 32;
        build_group_kernel<<<(unsigned)((tot + 255) / 256), 256, 0, stream>>>(pc, xyz1, point_feats, idxA, Nn, NP1, NSAMP, 3, 32, bufA);
        static const int dims0[4] = {6, 64, 64, 64};
        _Float16* H = run_stack(sa1b0, dims0, 3, cols1, bufA, bufB, nullptr, true);
        max_ns_kernel<<<(Bn * NP1 * 64 + 255) / 256, 256, 0, stream>>>(H, 64, NP1, NSAMP, 192, 0, f1, nullptr);
    }
    {   // branch 1: 6 -> 64 -> 64 -> 128
        size_t tot = (size_t)Bn * cols1 * 32;
        build_group_kernel<<<(unsigned)((tot + 255) / 256), 256, 0, stream>>>(pc, xyz1, point_feats, idxB, Nn, NP1, NSAMP, 3, 32, bufA);
        static const int dims1[4] = {6, 64, 64, 128};
        _Float16* H = run_stack(sa1b1, dims1, 3, cols1, bufA, bufB, nullptr, true);
        max_ns_kernel<<<(Bn * NP1 * 128 + 255) / 256, 256, 0, stream>>>(H, 128, NP1, NSAMP, 192, 64, f1, nullptr);
    }
    // ---- SA level 2 ----
    fps_kernel<<<Bn, 1024, 0, stream>>>(xyz1, NP1, NP2, idxf2);
    gather_xyz_kernel<<<(Bn * NP2 * 3 + 255) / 256, 256, 0, stream>>>(xyz1, idxf2, xyz2, NP1, NP2);
    {
        int blocks = (Bn * NP2 * 32 + 255) / 256;
        ball_query_kernel<<<blocks, 256, 0, stream>>>(xyz1, xyz2, NP1, NP2, 0.16f, NSAMP, idxC);
    }
    const int cols2 = NP2 * NSAMP;  // 32768
    {
        size_t tot = (size_t)Bn * cols2 * 224;    // Kp(195)=224
        build_group_kernel<<<(unsigned)((tot + 255) / 256), 256, 0, stream>>>(xyz1, xyz2, f1, idxC, NP1, NP2, NSAMP, 192, 224, bufA);
        static const int dims2[4] = {195, 128, 128, 256};
        _Float16* H = run_stack(sa2cg, dims2, 3, cols2, bufA, bufB, nullptr, true);
        max_ns_kernel<<<(Bn * NP2 * 256 + 255) / 256, 256, 0, stream>>>(H, 256, NP2, NSAMP, 256, 0, f2, f2h);
    }
    // ---- FP2: (xyz1 <- xyz2), 448 -> 256 -> 128 ----
    knn3_kernel<<<(Bn * NP1 + 255) / 256, 256, 0, stream>>>(xyz1, xyz2, NP1, NP2, idx3a, w3a);
    {
        size_t tot = (size_t)Bn * NP1 * 448;
        interp_concat_kernel<<<(unsigned)((tot + 255) / 256), 256, 0, stream>>>(f2, 256, NP2, f1, 192, NP1, idx3a, w3a, 448, bufA);
        static const int dimsf2[3] = {448, 256, 128};
        run_stack(fp2cg, dimsf2, 2, NP1, bufA, bufB, f1u, false);
    }
    // ---- FP1: (pc <- xyz1), 131 -> 128 -> 128 -> 64 ----
    knn3_kernel<<<(Bn * Nn + 255) / 256, 256, 0, stream>>>(pc, xyz1, Nn, NP1, idx3b, w3b);
    {
        size_t tot = (size_t)Bn * Nn * 160;       // Kp(131)=160
        interp_concat_kernel<<<(unsigned)((tot + 255) / 256), 256, 0, stream>>>(f1u, 128, NP1, point_feats, 3, Nn, idx3b, w3b, 160, bufA);
        static const int dimsf1[4] = {131, 128, 128, 64};
        run_stack(fp1cg, dimsf1, 3, Nn, bufA, bufB, f0u, false);
    }
    // ---- decoder memory: mem = f2 @ inW^T + inb, stored f16 (B,512,256) ----
    gemm_any(inW, f2h, memh, inb, 256, 256, 256, NP2, true);
    // ---- decoder ----
    bcast_query_kernel<<<(16 * 256 + 255) / 256, 256, 0, stream>>>(queryP, tgt);
    const int linB = (Bn * 16 * 256 + 255) / 256;
    for (int l = 0; l < 2; l++) {
        const LayerP& L = dec[l];
        // self-attention
        linear_kernel<<<linB, 256, 0, stream>>>(tgt, L.slf.Wq, L.slf.bq, tq, 16, 256, 256, 0);
        linear_kernel<<<linB, 256, 0, stream>>>(tgt, L.slf.Wk, L.slf.bk, tk, 16, 256, 256, 0);
        linear_kernel<<<linB, 256, 0, stream>>>(tgt, L.slf.Wv, L.slf.bv, tv, 16, 256, 256, 0);
        mha_kernel<<<Bn * 8 * 16, 32, 0, stream>>>(tq, tk, tv, 16, 16, ta);
        linear_kernel<<<linB, 256, 0, stream>>>(ta, L.slf.Wo, L.slf.bo, tb, 16, 256, 256, 0);
        add_ln_kernel<<<Bn * 16, 256, 0, stream>>>(tgt, tb, L.ln1g, L.ln1b);
        // cross-attention: K/V projections over 512 tokens via WMMA
        linear_kernel<<<linB, 256, 0, stream>>>(tgt, L.cross.Wq, L.cross.bq, tq, 16, 256, 256, 0);
        gemm_any(L.cross.Wk, memh, KT_, L.cross.bk, 256, 256, 256, NP2, false);
        gemm_any(L.cross.Wv, memh, VT_, L.cross.bv, 256, 256, 256, NP2, false);
        mha_kernel<<<Bn * 8 * 16, 32, 0, stream>>>(tq, KT_, VT_, 16, NP2, ta);
        linear_kernel<<<linB, 256, 0, stream>>>(ta, L.cross.Wo, L.cross.bo, tb, 16, 256, 256, 0);
        add_ln_kernel<<<Bn * 16, 256, 0, stream>>>(tgt, tb, L.ln2g, L.ln2b);
        // feed-forward
        linear_kernel<<<linB, 256, 0, stream>>>(tgt, L.W1, L.b1, ta, 16, 256, 256, 1);
        linear_kernel<<<linB, 256, 0, stream>>>(ta, L.W2, L.b2, tb, 16, 256, 256, 0);
        add_ln_kernel<<<Bn * 16, 256, 0, stream>>>(tgt, tb, L.ln3g, L.ln3b);
    }
    // ---- object head (token-major: no transpose needed) ----
    {
        size_t tot = (size_t)Bn * 16 * 256;
        f32_to_f16_kernel<<<(unsigned)((tot + 255) / 256), 256, 0, stream>>>(tgt, sTh, tot);
        static const int dimso[2] = {256, 256};
        _Float16* H = run_stack(&obj1cg, dimso, 1, 16, sTh, bufB, nullptr, true);
        gemm_any(obj2W, H, s2, obj2b, 64, 256, 256, 16, false);
    }
    // ---- logits + softmax ----
    logits_softmax_kernel<<<dim3((Nn + 255) / 256, Bn), 256, 0, stream>>>(f0u, s2, (float*)d_out, Nn);
}
